// LocalWinCrossAttnOcc_23072564314120
// MI455X (gfx1250) — matrix operand order repaired and checked by reading, not compiled, not run on hardware
//
#include <hip/hip_runtime.h>
#include <hip/hip_bf16.h>

typedef __attribute__((ext_vector_type(16))) _Float16 v16h;
typedef __attribute__((ext_vector_type(8)))  _Float16 v8h;
typedef __attribute__((ext_vector_type(4)))  _Float16 v4h;
typedef __attribute__((ext_vector_type(8)))  float    v8f;
typedef __attribute__((ext_vector_type(4)))  unsigned int v4u;
typedef __attribute__((ext_vector_type(8)))  int      v8i;
typedef __attribute__((ext_vector_type(4)))  int      v4i;

#define B_     2
#define CI     256
#define CR     128
#define HH     384
#define WW_    384
#define Hh     96
#define Ww     96
#define HWp    (Hh*Ww)          // 9216
#define D_     64
#define GROUPS 8
#define WIN    7
#define PADW   3
#define WINW   22               // 16 + 2*3
#define NWIN   (WINW*WINW)      // 484

// ---------------------------------------------------------------- utilities
__global__ __launch_bounds__(256) void cvt_f32_f16(const float* __restrict__ in,
                                                   _Float16* __restrict__ out, int n) {
    int i = blockIdx.x * 256 + threadIdx.x;
    if (i < n) out[i] = (_Float16)in[i];
}

// ---------------------------------------------------------------- pooling
// outputs are PIXEL-MAJOR: out[b][n][c] (f16) so GEMM A-fragments are contiguous.
__global__ __launch_bounds__(256) void pool_avg4(const float* __restrict__ in,
                                                 _Float16* __restrict__ out, int total) {
    int idx = blockIdx.x * 256 + threadIdx.x;
    if (idx >= total) return;
    int n = idx % HWp;
    int bc = idx / HWp;
    int c = bc % CI, b = bc / CI;
    int y = n / Ww, x = n % Ww;
    const float* p = in + ((size_t)bc * HH + y * 4) * WW_ + x * 4;
    float s = 0.f;
#pragma unroll
    for (int r = 0; r < 4; ++r) {
        float4 v = *(const float4*)(p + (size_t)r * WW_);
        s += v.x + v.y + v.z + v.w;
    }
    out[((size_t)b * HWp + n) * CI + c] = (_Float16)(s * (1.0f / 16.0f));
}

__global__ __launch_bounds__(256) void pool_max4(const float* __restrict__ in,
                                                 _Float16* __restrict__ out, int total) {
    int idx = blockIdx.x * 256 + threadIdx.x;
    if (idx >= total) return;
    int n = idx % HWp;
    int bc = idx / HWp;
    int c = bc % CR, b = bc / CR;
    int y = n / Ww, x = n % Ww;
    const float* p = in + ((size_t)bc * HH + y * 4) * WW_ + x * 4;
    float m = -3.4e38f;
#pragma unroll
    for (int r = 0; r < 4; ++r) {
        float4 v = *(const float4*)(p + (size_t)r * WW_);
        m = fmaxf(m, fmaxf(fmaxf(v.x, v.y), fmaxf(v.z, v.w)));
    }
    out[((size_t)b * HWp + n) * CR + c] = (_Float16)m;
}

__global__ __launch_bounds__(256) void pool_max4_f32(const float* __restrict__ in,
                                                     float* __restrict__ out, int total) {
    int idx = blockIdx.x * 256 + threadIdx.x;
    if (idx >= total) return;
    int n = idx % HWp;
    int b = idx / HWp;
    int y = n / Ww, x = n % Ww;
    const float* p = in + ((size_t)b * HH + y * 4) * WW_ + x * 4;
    float m = -3.4e38f;
#pragma unroll
    for (int r = 0; r < 4; ++r) {
        float4 v = *(const float4*)(p + (size_t)r * WW_);
        m = fmaxf(m, fmaxf(fmaxf(v.x, v.y), fmaxf(v.z, v.w)));
    }
    out[idx] = m;
}

// ---------------------------------------------------------------- TDM tile fetch
// 2D tile: 64 rows x 64B, row stride K*2 bytes, padded +16B per 64B row in LDS
// (stride 80B = 40 halfs -> conflict-free 16-lane b128 reads).
__device__ __forceinline__ void tdm_load_tile(unsigned lds_addr,
                                              unsigned long long gaddr, int K) {
    v4u g0;
    g0[0] = 1u;                                    // count=1, user mode
    g0[1] = lds_addr;                              // LDS byte address
    g0[2] = (unsigned)(gaddr & 0xffffffffull);     // global_addr lo
    g0[3] = (unsigned)(gaddr >> 32) | (2u << 30);  // global_addr hi | type=2
    unsigned t0 = (unsigned)(K / 4);               // row length in 8B units
    v8i g1;
    g1[0] = (int)((3u << 16)       // data_size = 8B
                | (1u << 20)       // pad_enable
                | (3u << 22)       // pad_interval: every 16 dwords (64B)
                | (3u << 25));     // pad_amount: 4 dwords (16B)
    g1[1] = (int)((t0 & 0xffffu) << 16);                          // tensor_dim0 lo16
    g1[2] = (int)((t0 >> 16) | ((unsigned)(HWp & 0xffff) << 16)); // dim0 hi | dim1 lo
    g1[3] = (int)(((unsigned)HWp >> 16) | (8u << 16));            // dim1 hi | tile_dim0=8
    g1[4] = 64;                                                   // tile_dim1=64 rows
    g1[5] = (int)t0;                                              // tensor_dim0_stride lo32
    g1[6] = 0;
    g1[7] = 0;
    v4i gz4 = {0, 0, 0, 0};
    v8i gz8 = {0, 0, 0, 0, 0, 0, 0, 0};
    __builtin_amdgcn_tensor_load_to_lds(g0, g1, gz4, gz4, gz8, 0);
}

// ---------------------------------------------------------------- WMMA GEMM
// Y[b][pix][ch] = Xt[b][pix][K] * Wt[ch][K]^T  (f16 in, f32 out),
// optional row gate *clip(occ[b][pix],0,1).
// block 256 thr = 8 waves; tile 64 pix x 32 ch; wave(wm,wn) -> 16x16 D tile.
// A panels double-buffered via TENSOR_LOAD_TO_LDS (TENSORcnt).
#define LDA 40
#define KMAX 256
__global__ __launch_bounds__(256) void gemm_wmma(const _Float16* __restrict__ Wt,
                                                 const _Float16* __restrict__ Xt,
                                                 float* __restrict__ Y,
                                                 const float* __restrict__ occ,
                                                 int Nout, int K) {
    __shared__ _Float16 Wlds[32 * (KMAX + 8)];
    __shared__ _Float16 Apan[2][64 * LDA];

    const int b    = blockIdx.z;
    const int m0   = blockIdx.x * 64;   // pixel tile
    const int n0   = blockIdx.y * 32;   // out-channel tile
    const int tid  = threadIdx.x;
    const int wave = tid >> 5;
    const int lane = tid & 31;
    const int wm = wave >> 1, wn = wave & 1;
    const int half = lane >> 4, l16 = lane & 15;
    const int ldw  = K + 8;

    // weight panel [32 x K] -> LDS, padded rows (bank spread), 16B vectors
    for (int idx = tid; idx < 32 * K / 8; idx += 256) {
        int ch = (idx * 8) / K;
        int k  = (idx * 8) % K;
        *(v8h*)&Wlds[ch * ldw + k] = *(const v8h*)(Wt + (size_t)(n0 + ch) * K + k);
    }

    const unsigned lds0 = (unsigned)(size_t)(&Apan[0][0]);
    const unsigned lds1 = (unsigned)(size_t)(&Apan[1][0]);
    const unsigned long long gbase =
        (unsigned long long)(size_t)(Xt + ((size_t)b * HWp + m0) * K);
    const int nk = K / 32;

    if (wave == 0) tdm_load_tile(lds0, gbase, K);

    v8f acc = {};
    for (int s = 0; s < nk; ++s) {
        if (wave == 0) {
            if (s + 1 < nk) {
                tdm_load_tile((s & 1) ? lds0 : lds1, gbase + (size_t)(s + 1) * 64, K);
                __builtin_amdgcn_s_wait_tensorcnt(1);   // oldest (current) panel done
            } else {
                __builtin_amdgcn_s_wait_tensorcnt(0);   // last panel done
            }
        }
        __syncthreads();   // publish panel s (and weights on s==0)

        const _Float16* Ab = Apan[s & 1];
        const int k0 = s * 32;
        v16h a, bf;
        const _Float16* ap = Ab + (wm * 16 + l16) * LDA + half * 8;
        v8h a0 = *(const v8h*)ap;
        v8h a1 = *(const v8h*)(ap + 16);
        const _Float16* bp = &Wlds[(wn * 16 + l16) * ldw + k0 + half * 8];
        v8h b0 = *(const v8h*)bp;
        v8h b1 = *(const v8h*)(bp + 16);
#pragma unroll
        for (int e = 0; e < 8; ++e) {
            a[e] = a0[e]; a[8 + e] = a1[e];
            bf[e] = b0[e]; bf[8 + e] = b1[e];
        }
        acc = __builtin_amdgcn_wmma_f32_16x16x32_f16(false, a, false, bf,
                                                     (short)0, acc, false, false);
        __syncthreads();   // all waves done with panel s -> buffer reusable
    }

    const int ch = n0 + wn * 16 + l16;
#pragma unroll
    for (int v = 0; v < 8; ++v) {
        int pix = m0 + wm * 16 + v + (half ? 8 : 0);
        float om = 1.0f;
        if (occ) om = fminf(fmaxf(occ[(size_t)b * HWp + pix], 0.0f), 1.0f);
        Y[((size_t)b * HWp + pix) * Nout + ch] = acc[v] * om;
    }
}

// ---------------------------------------------------------------- GroupNorm stats
// combo = ((t*B_) + b)*8 + g  (t: 0=q,1=k). gn[combo*2]=mu, gn[combo*2+1]=rsqrt(var+eps)
// tensors are pixel-major [B][HW][64].
__global__ __launch_bounds__(256) void gn_stats(const float* __restrict__ qraw,
                                                const float* __restrict__ kraw,
                                                float* __restrict__ gn) {
    __shared__ float sh1[256], sh2[256];
    int combo = blockIdx.x;
    int t = combo >> 4;
    int b = (combo >> 3) & 1;
    int g = combo & 7;
    const float* src = (t ? kraw : qraw) + (size_t)b * HWp * D_ + g * 8;
    float s = 0.f, s2 = 0.f;
    for (int i = threadIdx.x; i < HWp; i += 256) {
        const float* p = src + (size_t)i * D_;
        float4 a = *(const float4*)p;
        float4 c = *(const float4*)(p + 4);
        s  += a.x + a.y + a.z + a.w + c.x + c.y + c.z + c.w;
        s2 += a.x*a.x + a.y*a.y + a.z*a.z + a.w*a.w
            + c.x*c.x + c.y*c.y + c.z*c.z + c.w*c.w;
    }
    sh1[threadIdx.x] = s; sh2[threadIdx.x] = s2;
    __syncthreads();
    for (int st = 128; st > 0; st >>= 1) {
        if (threadIdx.x < st) {
            sh1[threadIdx.x] += sh1[threadIdx.x + st];
            sh2[threadIdx.x] += sh2[threadIdx.x + st];
        }
        __syncthreads();
    }
    if (threadIdx.x == 0) {
        const float cnt = 8.0f * HWp;
        float mu = sh1[0] / cnt;
        float var = sh2[0] / cnt - mu * mu;
        gn[combo * 2 + 0] = mu;
        gn[combo * 2 + 1] = rsqrtf(var + 1e-5f);
    }
}

// ---------------------------------------------------------------- window attention
// 16x16 pixel tile / block; k/v halo staged normalized-f16 in ~124KB dynamic LDS
// (CDNA5: 320KB/WGP). Pixel-major tensors -> fully vectorized staging & stores.
__global__ __launch_bounds__(256) void attn_win(const float* __restrict__ qraw,
                                                const float* __restrict__ kraw,
                                                const float* __restrict__ vraw,
                                                const float* __restrict__ occ_p,
                                                const float* __restrict__ gn,
                                                const float* __restrict__ gq_w,
                                                const float* __restrict__ gq_b,
                                                const float* __restrict__ gk_w,
                                                const float* __restrict__ gk_b,
                                                _Float16* __restrict__ msg16) {
    extern __shared__ char smem_raw[];
    _Float16* ks   = (_Float16*)smem_raw;           // [NWIN][64]
    _Float16* vs   = ks + NWIN * D_;                // [NWIN][64]
    float*    occs = (float*)(vs + NWIN * D_);      // [NWIN]

    const int b   = blockIdx.z;
    const int x0  = blockIdx.x * 16;
    const int y0  = blockIdx.y * 16;
    const int tid = threadIdx.x;

    for (int idx = tid; idx < NWIN * 16; idx += 256) {
        int wp = idx >> 4;
        int d0 = (idx & 15) * 4;
        int gy = y0 - PADW + wp / WINW;
        int gx = x0 - PADW + wp % WINW;
        v4h kh = {}, vh = {};
        if (gy >= 0 && gy < Hh && gx >= 0 && gx < Ww) {
            size_t base = ((size_t)b * HWp + gy * Ww + gx) * D_ + d0;
            float4 kf = *(const float4*)(kraw + base);
            float4 vf = *(const float4*)(vraw + base);
            int g = d0 >> 3;
            float mu  = gn[(16 + b * 8 + g) * 2 + 0];
            float rs_ = gn[(16 + b * 8 + g) * 2 + 1];
            kh[0] = (_Float16)((kf.x - mu) * rs_ * gk_w[d0+0] + gk_b[d0+0]);
            kh[1] = (_Float16)((kf.y - mu) * rs_ * gk_w[d0+1] + gk_b[d0+1]);
            kh[2] = (_Float16)((kf.z - mu) * rs_ * gk_w[d0+2] + gk_b[d0+2]);
            kh[3] = (_Float16)((kf.w - mu) * rs_ * gk_w[d0+3] + gk_b[d0+3]);
            vh[0] = (_Float16)vf.x; vh[1] = (_Float16)vf.y;
            vh[2] = (_Float16)vf.z; vh[3] = (_Float16)vf.w;
        }
        *(v4h*)&ks[wp * D_ + d0] = kh;
        *(v4h*)&vs[wp * D_ + d0] = vh;
    }
    for (int idx = tid; idx < NWIN; idx += 256) {
        int gy = y0 - PADW + idx / WINW;
        int gx = x0 - PADW + idx % WINW;
        occs[idx] = (gy >= 0 && gy < Hh && gx >= 0 && gx < Ww)
                        ? occ_p[(size_t)b * HWp + gy * Ww + gx] : 0.f;
    }
    __syncthreads();

    const int ty = tid >> 4, tx = tid & 15;
    const int py = y0 + ty, px = x0 + tx;
    const size_t pbase = ((size_t)b * HWp + py * Ww + px) * D_;

    _Float16 qreg[D_];
#pragma unroll
    for (int d0 = 0; d0 < D_; d0 += 4) {
        float4 qf = *(const float4*)(qraw + pbase + d0);
        int g = d0 >> 3;
        float mu  = gn[(b * 8 + g) * 2 + 0];
        float rs_ = gn[(b * 8 + g) * 2 + 1];
        qreg[d0+0] = (_Float16)((qf.x - mu) * rs_ * gq_w[d0+0] + gq_b[d0+0]);
        qreg[d0+1] = (_Float16)((qf.y - mu) * rs_ * gq_w[d0+1] + gq_b[d0+1]);
        qreg[d0+2] = (_Float16)((qf.z - mu) * rs_ * gq_w[d0+2] + gq_b[d0+2]);
        qreg[d0+3] = (_Float16)((qf.w - mu) * rs_ * gq_w[d0+3] + gq_b[d0+3]);
    }

    float logits[WIN * WIN];
    float mx = -1e30f;
#pragma unroll
    for (int i = 0; i < WIN; ++i) {
#pragma unroll
        for (int j = 0; j < WIN; ++j) {
            int wp = (ty + i) * WINW + (tx + j);
            const _Float16* kp = &ks[wp * D_];
            float dot = 0.f;
#pragma unroll
            for (int d = 0; d < D_; ++d)
                dot = fmaf((float)qreg[d], (float)kp[d], dot);
            float lg = dot * 0.125f + 2.0f * __logf(fmaxf(occs[wp], 1e-6f));
            logits[i * WIN + j] = lg;
            mx = fmaxf(mx, lg);
        }
    }
    float ssum = 0.f;
#pragma unroll
    for (int p = 0; p < WIN * WIN; ++p) {
        float e = __expf(logits[p] - mx);
        logits[p] = e;
        ssum += e;
    }
    const float inv = 1.0f / ssum;

    for (int d0 = 0; d0 < D_; d0 += 8) {
        float accv[8] = {0.f, 0.f, 0.f, 0.f, 0.f, 0.f, 0.f, 0.f};
#pragma unroll
        for (int i = 0; i < WIN; ++i) {
#pragma unroll
            for (int j = 0; j < WIN; ++j) {
                int wp = (ty + i) * WINW + (tx + j);
                float a = logits[i * WIN + j];
                const _Float16* vp = &vs[wp * D_ + d0];
#pragma unroll
                for (int u = 0; u < 8; ++u)
                    accv[u] = fmaf(a, (float)vp[u], accv[u]);
            }
        }
        v8h outv;
#pragma unroll
        for (int u = 0; u < 8; ++u) outv[u] = (_Float16)(accv[u] * inv);
        *(v8h*)&msg16[pbase + d0] = outv;   // contiguous 16B store
    }
}

// ---------------------------------------------------------------- upsample + residual
// msg_o is pixel-major [B][HW][CI]
__global__ __launch_bounds__(256) void upsample_add(const float* __restrict__ img_bev,
                                                    const float* __restrict__ msg_o,
                                                    const float* __restrict__ alpha_p,
                                                    float* __restrict__ out, size_t total) {
    size_t idx = (size_t)blockIdx.x * 256 + threadIdx.x;
    if (idx >= total) return;
    int x  = (int)(idx % WW_);
    int y  = (int)((idx / WW_) % HH);
    size_t bc = idx / ((size_t)WW_ * HH);
    int c = (int)(bc % CI);
    int b = (int)(bc / CI);

    float sy = (y + 0.5f) * 0.25f - 0.5f;
    float sx = (x + 0.5f) * 0.25f - 0.5f;
    int iy0 = (int)floorf(sy);
    int ix0 = (int)floorf(sx);
    float fy = sy - (float)iy0;
    float fx = sx - (float)ix0;
    int iy1 = iy0 + 1, ix1 = ix0 + 1;
    iy0 = max(0, min(Hh - 1, iy0)); iy1 = max(0, min(Hh - 1, iy1));
    ix0 = max(0, min(Ww - 1, ix0)); ix1 = max(0, min(Ww - 1, ix1));

    const float* mp = msg_o + (size_t)b * HWp * CI + c;
    float v00 = mp[(size_t)(iy0 * Ww + ix0) * CI], v01 = mp[(size_t)(iy0 * Ww + ix1) * CI];
    float v10 = mp[(size_t)(iy1 * Ww + ix0) * CI], v11 = mp[(size_t)(iy1 * Ww + ix1) * CI];
    float v = (1.f - fy) * ((1.f - fx) * v00 + fx * v01) + fy * ((1.f - fx) * v10 + fx * v11);
    out[idx] = img_bev[idx] + alpha_p[0] * v;
}

// ---------------------------------------------------------------- launch
extern "C" void kernel_launch(void* const* d_in, const int* in_sizes, int n_in,
                              void* d_out, int out_size, void* d_ws, size_t ws_size,
                              hipStream_t stream) {
    (void)in_sizes; (void)n_in; (void)out_size; (void)ws_size;
    const float* img_bev  = (const float*)d_in[0];
    const float* rad_bev  = (const float*)d_in[1];
    const float* occ_prob = (const float*)d_in[2];
    const float* Wq = (const float*)d_in[3];
    const float* Wk = (const float*)d_in[4];
    const float* Wv = (const float*)d_in[5];
    const float* Wo = (const float*)d_in[6];
    const float* gq_w = (const float*)d_in[7];
    const float* gq_b = (const float*)d_in[8];
    const float* gk_w = (const float*)d_in[9];
    const float* gk_b = (const float*)d_in[10];
    const float* alpha = (const float*)d_in[11];

    size_t off = 0;
    char* base = (char*)d_ws;
    auto alloc = [&](size_t bytes) -> char* {
        char* p = base + off;
        off = (off + bytes + 255) & ~(size_t)255;
        return p;
    };
    _Float16* wq16  = (_Float16*)alloc((size_t)D_ * CI * 2);
    _Float16* wk16  = (_Float16*)alloc((size_t)D_ * CR * 2);
    _Float16* wv16  = (_Float16*)alloc((size_t)D_ * CR * 2);
    _Float16* wo16  = (_Float16*)alloc((size_t)CI * D_ * 2);
    _Float16* img_p = (_Float16*)alloc((size_t)B_ * CI * HWp * 2);   // [B][HW][CI]
    _Float16* rad_p = (_Float16*)alloc((size_t)B_ * CR * HWp * 2);   // [B][HW][CR]
    float*    occ_p = (float*)alloc((size_t)B_ * HWp * 4);           // [B][HW]
    float*    qraw  = (float*)alloc((size_t)B_ * D_ * HWp * 4);      // [B][HW][64]
    float*    kraw  = (float*)alloc((size_t)B_ * D_ * HWp * 4);
    float*    vraw  = (float*)alloc((size_t)B_ * D_ * HWp * 4);
    float*    gn    = (float*)alloc(64 * 4);
    _Float16* msg16 = (_Float16*)alloc((size_t)B_ * D_ * HWp * 2);   // [B][HW][64]
    float*    msg_o = (float*)alloc((size_t)B_ * CI * HWp * 4);      // [B][HW][CI]

    // 1. weights to f16 (already [Nout][K] row-major)
    cvt_f32_f16<<<(D_ * CI + 255) / 256, 256, 0, stream>>>(Wq, wq16, D_ * CI);
    cvt_f32_f16<<<(D_ * CR + 255) / 256, 256, 0, stream>>>(Wk, wk16, D_ * CR);
    cvt_f32_f16<<<(D_ * CR + 255) / 256, 256, 0, stream>>>(Wv, wv16, D_ * CR);
    cvt_f32_f16<<<(CI * D_ + 255) / 256, 256, 0, stream>>>(Wo, wo16, CI * D_);

    // 2. pooling (pixel-major outputs)
    {
        int t_img = B_ * CI * HWp;
        int t_rad = B_ * CR * HWp;
        int t_occ = B_ * HWp;
        pool_avg4<<<(t_img + 255) / 256, 256, 0, stream>>>(img_bev, img_p, t_img);
        pool_max4<<<(t_rad + 255) / 256, 256, 0, stream>>>(rad_bev, rad_p, t_rad);
        pool_max4_f32<<<(t_occ + 255) / 256, 256, 0, stream>>>(occ_prob, occ_p, t_occ);
    }

    // 3. q/k/v projections (WMMA + TDM double-buffered A panels)
    {
        dim3 g(HWp / 64, D_ / 32, B_);
        gemm_wmma<<<g, 256, 0, stream>>>(wq16, img_p, qraw, nullptr, D_, CI);
        gemm_wmma<<<g, 256, 0, stream>>>(wk16, rad_p, kraw, nullptr, D_, CR);
        gemm_wmma<<<g, 256, 0, stream>>>(wv16, rad_p, vraw, nullptr, D_, CR);
    }

    // 4. GroupNorm stats (q and k)
    gn_stats<<<2 * B_ * GROUPS, 256, 0, stream>>>(qraw, kraw, gn);

    // 5. window attention
    {
        size_t shmem = (size_t)NWIN * D_ * 2 * sizeof(_Float16) + NWIN * sizeof(float);
        (void)hipFuncSetAttribute(reinterpret_cast<const void*>(attn_win),
                                  hipFuncAttributeMaxDynamicSharedMemorySize, (int)shmem);
        dim3 g(Ww / 16, Hh / 16, B_);
        attn_win<<<g, 256, shmem, stream>>>(qraw, kraw, vraw, occ_p, gn,
                                            gq_w, gq_b, gk_w, gk_b, msg16);
    }

    // 6. output projection (WMMA) with fused occ gating
    {
        dim3 g(HWp / 64, CI / 32, B_);
        gemm_wmma<<<g, 256, 0, stream>>>(wo16, msg16, msg_o, occ_p, CI, D_);
    }

    // 7. bilinear 4x upsample + residual
    {
        size_t total = (size_t)B_ * CI * HH * WW_;
        upsample_add<<<(unsigned)((total + 255) / 256), 256, 0, stream>>>(
            img_bev, msg_o, alpha, (float*)d_out, total);
    }
}
